// MultiHeadAttention_68058051772548
// MI455X (gfx1250) — matrix operand order repaired and checked by reading, not compiled, not run on hardware
//
#include <hip/hip_runtime.h>
#include <math.h>

// Problem constants (match reference)
constexpr int Bc = 4;
constexpr int Sc = 2048;
constexpr int Ec = 1024;
constexpr int Hc = 16;
constexpr int Dc = 64;
constexpr int HDc = Hc * Dc;   // 1024

// ---------------- WMMA types ----------------
typedef __attribute__((ext_vector_type(16))) __bf16   v16bf;
typedef __attribute__((ext_vector_type(8)))  float    v8f;
typedef __attribute__((ext_vector_type(8)))  unsigned int v8u;

union FragU {
    v8u   u;
    v16bf bf;
};

__device__ __forceinline__ v8f wmma_bf16(const FragU& a, const FragU& b, v8f c) {
    return __builtin_amdgcn_wmma_f32_16x16x32_bf16(
        /*neg_a=*/false, a.bf, /*neg_b=*/false, b.bf,
        /*c_mod=*/(short)0, c, /*reuse_a=*/false, /*reuse_b=*/false);
}

// K-offset (bf16 elements) of dword j within a row for the A fragment (16x32 bf16)
__device__ __forceinline__ int koffA(int j, int half) {
    return 2 * j + 16 * half;
}
// K-offset of dword j within a column for the B fragment (32x16 bf16)
__device__ __forceinline__ int koffB(int j, int half) {
    return 2 * j + 16 * half;
}

__device__ __forceinline__ void loadA(FragU& f, const unsigned short* row, int half) {
#pragma unroll
    for (int j = 0; j < 8; ++j)
        f.u[j] = *reinterpret_cast<const unsigned int*>(row + koffA(j, half));
}
__device__ __forceinline__ void loadB(FragU& f, const unsigned short* col, int half) {
#pragma unroll
    for (int j = 0; j < 8; ++j)
        f.u[j] = *reinterpret_cast<const unsigned int*>(col + koffB(j, half));
}

__device__ __forceinline__ unsigned short f2bf(float f) {
    unsigned int u = __float_as_uint(f);
    u += 0x7FFFu + ((u >> 16) & 1u);   // round-to-nearest-even
    return (unsigned short)(u >> 16);
}
__device__ __forceinline__ unsigned int pack2bf(float a, float b) {
    return (unsigned int)f2bf(a) | ((unsigned int)f2bf(b) << 16);
}

// ---------------- conversion / transpose kernels ----------------
__global__ void k_cvt_bf16(const float* __restrict__ in,
                           unsigned short* __restrict__ out, int n2) {
    int i = blockIdx.x * blockDim.x + threadIdx.x;
    if (i < n2)
        reinterpret_cast<unsigned int*>(out)[i] = pack2bf(in[2 * i], in[2 * i + 1]);
}

// w: [H,E,D] f32 -> wT: [H,D,E] bf16
__global__ void k_wT(const float* __restrict__ w, unsigned short* __restrict__ wT) {
    int i = blockIdx.x * blockDim.x + threadIdx.x;   // over H*D*E, e fastest
    int e = i % Ec;
    int t = i / Ec;
    int d = t % Dc;
    int h = t / Dc;
    wT[i] = f2bf(w[((size_t)h * Ec + e) * Dc + d]);
}

// wo: [HD,E] f32 -> woT: [E,HD] bf16
__global__ void k_woT(const float* __restrict__ wo, unsigned short* __restrict__ woT) {
    int i = blockIdx.x * blockDim.x + threadIdx.x;   // over E*HD, c fastest
    int c = i % HDc;
    int e = i / HDc;
    woT[i] = f2bf(wo[(size_t)c * Ec + e]);
}

// ---------------- QKV projection ----------------
// out = (x[b] @ w[h] + bias[h]) * scale
// Each wave computes a 32x64 output tile: 2 A-frags + 4 B-frags -> 8 WMMAs / K-step.
// xb:  [B,S,E] bf16 ; wT: [H,D,E] bf16 ; bias: [H,D] f32
// transposed==0 : out [B,H,S,D] bf16 (q,k);  transposed==1 : out [B,H,D,S] bf16 (v)
__global__ void __launch_bounds__(256)
k_proj(const unsigned short* __restrict__ xb,
       const unsigned short* __restrict__ wT,
       const float* __restrict__ bias,
       unsigned short* __restrict__ out,
       float scale, int transposed) {
    const int lane = threadIdx.x & 31;
    const int wave = threadIdx.x >> 5;
    const int m    = lane & 15;
    const int half = lane >> 4;

    const int b  = blockIdx.z;
    const int h  = blockIdx.y;
    const int s0 = blockIdx.x * 256 + wave * 32;   // 32-row tile per wave

    const unsigned short* xrow0 = xb + ((size_t)b * Sc + s0 + m) * Ec;
    const unsigned short* xrow1 = xrow0 + (size_t)16 * Ec;
    const unsigned short* wcol  = wT + ((size_t)h * Dc + m) * Ec;   // + ct*16*E per col tile

    v8f acc[2][4];
#pragma unroll
    for (int rt = 0; rt < 2; ++rt)
#pragma unroll
        for (int ct = 0; ct < 4; ++ct) acc[rt][ct] = v8f{};

    for (int e0 = 0; e0 < Ec; e0 += 32) {
        // speculative prefetch of the streamed A rows (OOB at the tail is
        // silently dropped per ISA 10.5; workspace extends past the row)
        __builtin_prefetch(xrow0 + e0 + 256, 0, 1);   // global_prefetch_b8
        __builtin_prefetch(xrow1 + e0 + 256, 0, 1);

        FragU a0, a1, bb[4];
        loadA(a0, xrow0 + e0, half);
        loadA(a1, xrow1 + e0, half);
#pragma unroll
        for (int ct = 0; ct < 4; ++ct)
            loadB(bb[ct], wcol + (size_t)ct * 16 * Ec + e0, half);
#pragma unroll
        for (int ct = 0; ct < 4; ++ct) {
            acc[0][ct] = wmma_bf16(a0, bb[ct], acc[0][ct]);
            acc[1][ct] = wmma_bf16(a1, bb[ct], acc[1][ct]);
        }
    }

#pragma unroll
    for (int ct = 0; ct < 4; ++ct) {
        const float bv = bias[h * Dc + ct * 16 + m];
#pragma unroll
        for (int rt = 0; rt < 2; ++rt)
#pragma unroll
            for (int r = 0; r < 8; ++r) acc[rt][ct][r] = (acc[rt][ct][r] + bv) * scale;
    }

    if (!transposed) {
#pragma unroll
        for (int rt = 0; rt < 2; ++rt)
#pragma unroll
            for (int r = 0; r < 8; ++r) {
                int s = s0 + rt * 16 + r + 8 * half;
                unsigned short* orow = out + (((size_t)b * Hc + h) * Sc + s) * Dc;
#pragma unroll
                for (int ct = 0; ct < 4; ++ct)
                    orow[ct * 16 + m] = f2bf(acc[rt][ct][r]);
            }
    } else {
#pragma unroll
        for (int rt = 0; rt < 2; ++rt)
#pragma unroll
            for (int ct = 0; ct < 4; ++ct) {
                uint4 v4;
                v4.x = pack2bf(acc[rt][ct][0], acc[rt][ct][1]);
                v4.y = pack2bf(acc[rt][ct][2], acc[rt][ct][3]);
                v4.z = pack2bf(acc[rt][ct][4], acc[rt][ct][5]);
                v4.w = pack2bf(acc[rt][ct][6], acc[rt][ct][7]);
                size_t base = (((size_t)b * Hc + h) * Dc + ct * 16 + m) * Sc +
                              s0 + rt * 16 + 8 * half;
                *reinterpret_cast<uint4*>(out + base) = v4;
            }
    }
}

// ---------------- flash attention (causal) ----------------
// Each wave owns 32 query rows; K/V fragments are reused across both row tiles.
// q,k: [B,H,S,D] bf16 (q pre-scaled by 1/sqrt(D)); vT: [B,H,D,S] bf16
// o: [B,S,H*D] bf16
__global__ void __launch_bounds__(256)
k_attn(const unsigned short* __restrict__ q,
       const unsigned short* __restrict__ k,
       const unsigned short* __restrict__ vT,
       unsigned short* __restrict__ o) {
    __shared__ unsigned short plds[8][2][16 * 32];   // per-wave P staging (C->A layout)

    const int lane = threadIdx.x & 31;
    const int wave = threadIdx.x >> 5;
    const int m    = lane & 15;
    const int half = lane >> 4;

    const int b  = blockIdx.z;
    const int h  = blockIdx.y;
    const int q0 = blockIdx.x * 256 + wave * 32;

    const size_t bh = (size_t)b * Hc + h;
    const unsigned short* kp = k  + bh * Sc * Dc;
    const unsigned short* vp = vT + bh * Dc * Sc;

    // Q A-fragments: [row tile][K half], resident for the whole loop
    FragU aq[2][2];
#pragma unroll
    for (int rt = 0; rt < 2; ++rt) {
        const unsigned short* qp = q + (bh * Sc + q0 + rt * 16 + m) * Dc;
#pragma unroll
        for (int kk = 0; kk < 2; ++kk)
            loadA(aq[rt][kk], qp + kk * 32, half);
    }

    float mrow[2][8], lrow[2][8];
    v8f acc[2][4];
#pragma unroll
    for (int rt = 0; rt < 2; ++rt) {
#pragma unroll
        for (int r = 0; r < 8; ++r) { mrow[rt][r] = -3.0e38f; lrow[rt][r] = 0.0f; }
#pragma unroll
        for (int dt = 0; dt < 4; ++dt) acc[rt][dt] = v8f{};
    }

    for (int t0 = 0; t0 < q0 + 32; t0 += 32) {
        // ---- scores S = Q K^T : 2 row tiles x 2 col groups
        v8f sc[2][2];
#pragma unroll
        for (int rt = 0; rt < 2; ++rt)
#pragma unroll
            for (int g = 0; g < 2; ++g) sc[rt][g] = v8f{};

#pragma unroll
        for (int g = 0; g < 2; ++g) {
            const unsigned short* krow = kp + (size_t)(t0 + g * 16 + m) * Dc;
            FragU bk0, bk1;
            loadB(bk0, krow, half);
            loadB(bk1, krow + 32, half);
#pragma unroll
            for (int rt = 0; rt < 2; ++rt) {
                sc[rt][g] = wmma_bf16(aq[rt][0], bk0, sc[rt][g]);
                sc[rt][g] = wmma_bf16(aq[rt][1], bk1, sc[rt][g]);
            }
        }

        // ---- causal mask on diagonal tiles
        if (t0 + 31 > q0) {
#pragma unroll
            for (int rt = 0; rt < 2; ++rt)
#pragma unroll
                for (int r = 0; r < 8; ++r) {
                    int row = q0 + rt * 16 + r + 8 * half;
                    if (t0 + m > row)       sc[rt][0][r] = -3.0e38f;
                    if (t0 + 16 + m > row)  sc[rt][1][r] = -3.0e38f;
                }
        }

        // ---- online softmax (xor masks <16 stay within 16-lane halves)
#pragma unroll
        for (int rt = 0; rt < 2; ++rt)
#pragma unroll
            for (int r = 0; r < 8; ++r) {
                float tm = fmaxf(sc[rt][0][r], sc[rt][1][r]);
#pragma unroll
                for (int off = 8; off >= 1; off >>= 1) tm = fmaxf(tm, __shfl_xor(tm, off));
                float mn = fmaxf(mrow[rt][r], tm);
                float ef = __expf(mrow[rt][r] - mn);
                mrow[rt][r] = mn;
                sc[rt][0][r] = __expf(sc[rt][0][r] - mn);
                sc[rt][1][r] = __expf(sc[rt][1][r] - mn);
                float rs = sc[rt][0][r] + sc[rt][1][r];
#pragma unroll
                for (int off = 8; off >= 1; off >>= 1) rs += __shfl_xor(rs, off);
                lrow[rt][r] = lrow[rt][r] * ef + rs;
                acc[rt][0][r] *= ef; acc[rt][1][r] *= ef;
                acc[rt][2][r] *= ef; acc[rt][3][r] *= ef;
            }

        // ---- P (C layout) -> LDS as row-major 16x32 bf16 per row tile
#pragma unroll
        for (int rt = 0; rt < 2; ++rt) {
            unsigned short* pw = &plds[wave][rt][0];
#pragma unroll
            for (int r = 0; r < 8; ++r) {
                int row = r + 8 * half;
                pw[row * 32 + m]      = f2bf(sc[rt][0][r]);
                pw[row * 32 + 16 + m] = f2bf(sc[rt][1][r]);
            }
        }
        asm volatile("s_wait_dscnt 0" ::: "memory");   // per-wave RAW safety

        // ---- P back as A fragments
        FragU ap[2];
#pragma unroll
        for (int rt = 0; rt < 2; ++rt) {
            const unsigned int* pr =
                reinterpret_cast<const unsigned int*>(&plds[wave][rt][0] + m * 32);
#pragma unroll
            for (int j = 0; j < 8; ++j) ap[rt].u[j] = pr[koffA(j, half) >> 1];
        }

        // ---- O += P V : 4 d-tiles, V fragment reused by both row tiles
#pragma unroll
        for (int dt = 0; dt < 4; ++dt) {
            FragU bv;
            loadB(bv, vp + (size_t)(dt * 16 + m) * Sc + t0, half);
            acc[0][dt] = wmma_bf16(ap[0], bv, acc[0][dt]);
            acc[1][dt] = wmma_bf16(ap[1], bv, acc[1][dt]);
        }
    }

    // ---- epilogue: normalize and store concat-head output (bf16)
#pragma unroll
    for (int rt = 0; rt < 2; ++rt)
#pragma unroll
        for (int r = 0; r < 8; ++r) {
            float inv = 1.0f / lrow[rt][r];
            int s = q0 + rt * 16 + r + 8 * half;
            unsigned short* orow = o + ((size_t)b * Sc + s) * HDc + h * Dc;
#pragma unroll
            for (int dt = 0; dt < 4; ++dt)
                orow[dt * 16 + m] = f2bf(acc[rt][dt][r] * inv);
        }
}

// ---------------- output projection ----------------
// y[B*S, E] = o[B*S, HD] @ wo[HD, E] + bo ;  woT: [E, HD] bf16
// Each wave computes a 32x64 tile (2 A-frags x 4 B-frags -> 8 WMMAs / K-step).
__global__ void __launch_bounds__(256)
k_outproj(const unsigned short* __restrict__ ob,
          const unsigned short* __restrict__ woT,
          const float* __restrict__ bo,
          float* __restrict__ y) {
    const int lane = threadIdx.x & 31;
    const int wave = threadIdx.x >> 5;
    const int m    = lane & 15;
    const int half = lane >> 4;

    const int row0 = blockIdx.x * 64 + (wave >> 2) * 32;
    const int n0   = blockIdx.y * 256 + (wave & 3) * 64;

    const unsigned short* arow0 = ob  + (size_t)(row0 + m) * HDc;
    const unsigned short* arow1 = arow0 + (size_t)16 * HDc;
    const unsigned short* bcol  = woT + (size_t)(n0 + m) * HDc;   // + ct*16*HD per tile

    v8f acc[2][4];
#pragma unroll
    for (int rt = 0; rt < 2; ++rt)
#pragma unroll
        for (int ct = 0; ct < 4; ++ct) acc[rt][ct] = v8f{};

    for (int k0 = 0; k0 < HDc; k0 += 32) {
        __builtin_prefetch(arow0 + k0 + 256, 0, 1);   // speculative stream prefetch
        __builtin_prefetch(arow1 + k0 + 256, 0, 1);

        FragU a0, a1, bb[4];
        loadA(a0, arow0 + k0, half);
        loadA(a1, arow1 + k0, half);
#pragma unroll
        for (int ct = 0; ct < 4; ++ct)
            loadB(bb[ct], bcol + (size_t)ct * 16 * HDc + k0, half);
#pragma unroll
        for (int ct = 0; ct < 4; ++ct) {
            acc[0][ct] = wmma_bf16(a0, bb[ct], acc[0][ct]);
            acc[1][ct] = wmma_bf16(a1, bb[ct], acc[1][ct]);
        }
    }

#pragma unroll
    for (int ct = 0; ct < 4; ++ct) {
        const float bias = bo[n0 + ct * 16 + m];
#pragma unroll
        for (int rt = 0; rt < 2; ++rt)
#pragma unroll
            for (int r = 0; r < 8; ++r) {
                int row = row0 + rt * 16 + r + 8 * half;
                y[(size_t)row * Ec + n0 + ct * 16 + m] = acc[rt][ct][r] + bias;
            }
    }
}

// ---------------- host launch ----------------
extern "C" void kernel_launch(void* const* d_in, const int* in_sizes, int n_in,
                              void* d_out, int out_size, void* d_ws, size_t ws_size,
                              hipStream_t stream) {
    (void)in_sizes; (void)n_in; (void)out_size; (void)ws_size;

    const float* x  = (const float*)d_in[0];
    const float* wq = (const float*)d_in[1];
    const float* bq = (const float*)d_in[2];
    const float* wk = (const float*)d_in[3];
    const float* bk = (const float*)d_in[4];
    const float* wv = (const float*)d_in[5];
    const float* bv = (const float*)d_in[6];
    const float* wo = (const float*)d_in[7];
    const float* bo = (const float*)d_in[8];
    float* y = (float*)d_out;

    // workspace carving (256B aligned)
    unsigned char* wsp = (unsigned char*)d_ws;
    auto carve = [&](size_t bytes) -> void* {
        void* p = wsp;
        wsp += (bytes + 255) & ~(size_t)255;
        return p;
    };
    unsigned short* xb   = (unsigned short*)carve((size_t)Bc * Sc * Ec * 2);
    unsigned short* wqT  = (unsigned short*)carve((size_t)Hc * Dc * Ec * 2);
    unsigned short* wkT  = (unsigned short*)carve((size_t)Hc * Dc * Ec * 2);
    unsigned short* wvT  = (unsigned short*)carve((size_t)Hc * Dc * Ec * 2);
    unsigned short* woT  = (unsigned short*)carve((size_t)Ec * HDc * 2);
    unsigned short* qb   = (unsigned short*)carve((size_t)Bc * Hc * Sc * Dc * 2);
    unsigned short* kb   = (unsigned short*)carve((size_t)Bc * Hc * Sc * Dc * 2);
    unsigned short* vTb  = (unsigned short*)carve((size_t)Bc * Hc * Dc * Sc * 2);
    unsigned short* obuf = (unsigned short*)carve((size_t)Bc * Sc * HDc * 2);

    const int TB = 256;

    // 1) x -> bf16
    {
        int n2 = Bc * Sc * Ec / 2;
        k_cvt_bf16<<<n2 / TB, TB, 0, stream>>>(x, xb, n2);
    }
    // 2) weights -> transposed bf16
    {
        int n = Hc * Dc * Ec;
        k_wT<<<n / TB, TB, 0, stream>>>(wq, wqT);
        k_wT<<<n / TB, TB, 0, stream>>>(wk, wkT);
        k_wT<<<n / TB, TB, 0, stream>>>(wv, wvT);
        k_woT<<<(Ec * HDc) / TB, TB, 0, stream>>>(wo, woT);
    }
    // 3) QKV projections (scale 1/sqrt(D) folded into Q)
    {
        dim3 grid(Sc / 256, Hc, Bc);
        const float qscale = 1.0f / sqrtf((float)Dc);
        k_proj<<<grid, TB, 0, stream>>>(xb, wqT, bq, qb, qscale, 0);
        k_proj<<<grid, TB, 0, stream>>>(xb, wkT, bk, kb, 1.0f, 0);
        k_proj<<<grid, TB, 0, stream>>>(xb, wvT, bv, vTb, 1.0f, 1);
    }
    // 4) causal flash attention (32 query rows per wave)
    {
        dim3 grid(Sc / 256, Hc, Bc);
        k_attn<<<grid, TB, 0, stream>>>(qb, kb, vTb, obuf);
    }
    // 5) output projection
    {
        dim3 grid((Bc * Sc) / 64, Ec / 256);
        k_outproj<<<grid, TB, 0, stream>>>(obuf, woT, bo, y);
    }
}
